// HierarchicalReasoningModel_ACTV1MoEBlock_79611513799335
// MI455X (gfx1250) — matrix operand order repaired and checked by reading, not compiled, not run on hardware
//
#include <hip/hip_runtime.h>
#include <hip/hip_bf16.h>
#include <cstdint>

typedef __attribute__((ext_vector_type(16))) __bf16 v16bf;
typedef __attribute__((ext_vector_type(8)))  float  v8f;

namespace {

constexpr int kBch = 2;
constexpr int kSeq = 1024;
constexpr int kHid = 1024;
constexpr int kTok = kBch * kSeq;    // 2048
constexpr int kNH  = 16;
constexpr int kHD  = 64;
constexpr int kE   = 8;
constexpr int kInter = 2816;
constexpr int kGU    = 2 * kInter;   // 5632
constexpr float kEps = 1e-5f;

constexpr int BM = 128, BN = 64, BK = 32, LP = 8;   // LP: LDS row pad (bf16 elems)

union AFrag { v16bf v; unsigned u[8]; };
union CFrag { v8f   v; float    f[8]; };

__device__ __forceinline__ v8f zero_v8f() {
  CFrag z;
  #pragma unroll
  for (int i = 0; i < 8; ++i) z.f[i] = 0.0f;
  return z.v;
}

// ---------------------------------------------------------------------------
// CDNA5 async global->LDS copy (ASYNCcnt tracked), 16B per lane.
// GV addressing: vdst = LDS byte address (addr[31:0] of shared pointer),
// vaddr = 64-bit global address, no saddr ("off").
// ---------------------------------------------------------------------------
__device__ __forceinline__ void async_copy_b128(void* lds_dst, const void* gsrc) {
  unsigned lds = (unsigned)(size_t)lds_dst;
  unsigned long long ga = (unsigned long long)(size_t)gsrc;
  asm volatile("global_load_async_to_lds_b128 %0, %1, off"
               :: "v"(lds), "v"(ga) : "memory");
}
__device__ __forceinline__ void async_wait0() {
  asm volatile("s_wait_asynccnt 0x0" ::: "memory");
}

// ---------------------------------------------------------------------------
// Tile staging helpers (256-thread blocks).
// A tile: BMx32 bf16 via async-to-LDS (2 x b128 per thread).
// B tile: 32xBN bf16, transposed; staged through VGPRs for double buffering.
// ---------------------------------------------------------------------------
struct BStage { __bf16 r[8]; };

__device__ __forceinline__ void load_a_async_tile(const __bf16* __restrict__ A,
                                                  int Kd, int row0, int kb,
                                                  __bf16 (*As)[BK + LP]) {
  const int tid = threadIdx.x;
  #pragma unroll
  for (int i = 0; i < 2; ++i) {
    int idx = tid + i * 256;
    int r = idx >> 2, c4 = idx & 3;
    async_copy_b128(&As[r][c4 * 8], A + (size_t)(row0 + r) * Kd + kb + c4 * 8);
  }
}

__device__ __forceinline__ void load_b_regs(const __bf16* __restrict__ Bm, int N,
                                            int col0, int kb, BStage& st) {
  #pragma unroll
  for (int i = 0; i < 8; ++i) {
    int idx = threadIdx.x + i * 256;
    int k = idx >> 6, n = idx & 63;
    st.r[i] = Bm[(size_t)(kb + k) * N + col0 + n];
  }
}

__device__ __forceinline__ void store_b_regs(__bf16 (*Bt)[BK + LP], const BStage& st) {
  #pragma unroll
  for (int i = 0; i < 8; ++i) {
    int idx = threadIdx.x + i * 256;
    int k = idx >> 6, n = idx & 63;
    Bt[n][k] = st.r[i];
  }
}

__device__ __forceinline__ AFrag load_a_frag(const __bf16 (*As)[BK + LP],
                                             int m0, int half, int mr) {
  AFrag a;
  #pragma unroll
  for (int v = 0; v < 8; ++v) {
    int kk = ((v & 4) << 2) + half * 8 + (v & 3) * 2;
    a.u[v] = *(const unsigned*)&As[m0 + mr][kk];
  }
  return a;
}

__device__ __forceinline__ void wmma_row(const AFrag& a, const __bf16 (*Bt)[BK + LP],
                                         int half, int mr, v8f acc[4]) {
  AFrag b[4];
  #pragma unroll
  for (int nt = 0; nt < 4; ++nt)
    #pragma unroll
    for (int v = 0; v < 8; ++v)
      b[nt].u[v] = *(const unsigned*)&Bt[nt * 16 + mr][((v & 4) << 2) + half * 8 + (v & 3) * 2];
  #pragma unroll
  for (int nt = 0; nt < 4; ++nt)
    acc[nt] = __builtin_amdgcn_wmma_f32_16x16x32_bf16(
        false, a.v, false, b[nt].v, (short)0, acc[nt], false, false);
}

// ---------------------------------------------------------------------------
// Double-buffered, async-pipelined BF16 GEMM core: 256 threads (8 waves),
// block tile 128x64, K step 32. Async A-copy + VGPR-staged B for step s+1
// overlap the WMMA chain of step s.
// ---------------------------------------------------------------------------
__device__ __forceinline__ void gemm_core_pipe(
    const __bf16* __restrict__ A, const __bf16* __restrict__ Bm,
    int Kd, int N, int row0, int col0,
    __bf16 (*As2)[BM][BK + LP], __bf16 (*Bt2)[BN][BK + LP],
    v8f acc[4])
{
  const int lane = threadIdx.x & 31;
  const int half = lane >> 4;
  const int mr   = lane & 15;
  const int m0   = (threadIdx.x >> 5) * 16;

  #pragma unroll
  for (int i = 0; i < 4; ++i) acc[i] = zero_v8f();

  BStage bs;
  load_a_async_tile(A, Kd, row0, 0, As2[0]);
  load_b_regs(Bm, N, col0, 0, bs);
  store_b_regs(Bt2[0], bs);
  async_wait0();
  __syncthreads();

  const int nsteps = Kd / BK;
  for (int s = 0; s < nsteps; ++s) {
    const int cur = s & 1, nxt = cur ^ 1;
    const bool more = (s + 1) < nsteps;
    if (more) {
      load_a_async_tile(A, Kd, row0, (s + 1) * BK, As2[nxt]);  // overlaps WMMA
      load_b_regs(Bm, N, col0, (s + 1) * BK, bs);
    }
    AFrag a = load_a_frag(As2[cur], m0, half, mr);
    wmma_row(a, Bt2[cur], half, mr, acc);
    if (more) store_b_regs(Bt2[nxt], bs);
    async_wait0();
    __syncthreads();
  }
}

// MODE 0: C = acc ; MODE 1: C = acc + resid ; MODE 2: C += rowscale[row]*acc
template <int MODE>
__global__ __launch_bounds__(256)
void gemm_bf16_kernel(const __bf16* __restrict__ A, const __bf16* __restrict__ Bm,
                      float* __restrict__ C, int N, int Kd,
                      const float* __restrict__ resid,
                      const float* __restrict__ rowscale, int rsStride)
{
  __shared__ __align__(16) __bf16 As2[2][BM][BK + LP];
  __shared__ __align__(16) __bf16 Bt2[2][BN][BK + LP];
  const int lane = threadIdx.x & 31;
  const int half = lane >> 4;
  const int mr   = lane & 15;
  const int m0   = (threadIdx.x >> 5) * 16;
  const int row0 = blockIdx.x * BM;
  const int col0 = blockIdx.y * BN;

  v8f acc[4];
  gemm_core_pipe(A, Bm, Kd, N, row0, col0, As2, Bt2, acc);

  #pragma unroll
  for (int nt = 0; nt < 4; ++nt) {
    CFrag c; c.v = acc[nt];
    #pragma unroll
    for (int v = 0; v < 8; ++v) {
      int row = row0 + m0 + v + 8 * half;
      int col = col0 + nt * 16 + mr;
      size_t o = (size_t)row * N + col;
      if (MODE == 0)      C[o] = c.f[v];
      else if (MODE == 1) C[o] = c.f[v] + resid[o];
      else                C[o] += rowscale[(size_t)row * rsStride] * c.f[v];
    }
  }
}

// ---------------------------------------------------------------------------
// Fused MoE gate_up GEMM + SiLU*up epilogue, double-buffered. One block
// computes the gate tile (cols [col0,col0+64)) and matching up tile (+kInter)
// sharing one A tile; writes act bf16 [T, kInter]. No fp32 gu round-trip.
// ---------------------------------------------------------------------------
__global__ __launch_bounds__(256)
void moe_gateup_silu_kernel(const __bf16* __restrict__ A,
                            const __bf16* __restrict__ Bm,   // [kHid, kGU]
                            __bf16* __restrict__ act)        // [kTok, kInter]
{
  __shared__ __align__(16) __bf16 As2[2][BM][BK + LP];
  __shared__ __align__(16) __bf16 BtG[2][BN][BK + LP];
  __shared__ __align__(16) __bf16 BtU[2][BN][BK + LP];
  const int lane = threadIdx.x & 31;
  const int half = lane >> 4;
  const int mr   = lane & 15;
  const int m0   = (threadIdx.x >> 5) * 16;
  const int row0 = blockIdx.x * BM;
  const int col0 = blockIdx.y * BN;

  v8f gacc[4], uacc[4];
  #pragma unroll
  for (int i = 0; i < 4; ++i) { gacc[i] = zero_v8f(); uacc[i] = zero_v8f(); }

  BStage bg, bu;
  load_a_async_tile(A, kHid, row0, 0, As2[0]);
  load_b_regs(Bm, kGU, col0, 0, bg);
  load_b_regs(Bm, kGU, kInter + col0, 0, bu);
  store_b_regs(BtG[0], bg);
  store_b_regs(BtU[0], bu);
  async_wait0();
  __syncthreads();

  const int nsteps = kHid / BK;
  for (int s = 0; s < nsteps; ++s) {
    const int cur = s & 1, nxt = cur ^ 1;
    const bool more = (s + 1) < nsteps;
    if (more) {
      load_a_async_tile(A, kHid, row0, (s + 1) * BK, As2[nxt]);
      load_b_regs(Bm, kGU, col0, (s + 1) * BK, bg);
      load_b_regs(Bm, kGU, kInter + col0, (s + 1) * BK, bu);
    }
    AFrag a = load_a_frag(As2[cur], m0, half, mr);
    wmma_row(a, BtG[cur], half, mr, gacc);
    wmma_row(a, BtU[cur], half, mr, uacc);
    if (more) { store_b_regs(BtG[nxt], bg); store_b_regs(BtU[nxt], bu); }
    async_wait0();
    __syncthreads();
  }

  #pragma unroll
  for (int nt = 0; nt < 4; ++nt) {
    CFrag g, u; g.v = gacc[nt]; u.v = uacc[nt];
    #pragma unroll
    for (int v = 0; v < 8; ++v) {
      int row = row0 + m0 + v + 8 * half;
      int col = col0 + nt * 16 + mr;
      float gv = g.f[v];
      float silu = gv * __builtin_amdgcn_rcpf(1.0f + __expf(-gv));  // v_rcp_f32
      act[(size_t)row * kInter + col] = (__bf16)(silu * u.f[v]);
    }
  }
}

// ---------------------------------------------------------------------------
// QKV GEMM with fused RoPE. N-block of 64 == one head; rotate_half partner of
// accumulator tile nt lives in tile nt^2, same lane/VGPR.
// Outputs q/k/v in [B][NH][S][HD] bf16.
// ---------------------------------------------------------------------------
__global__ __launch_bounds__(256)
void qkv_rope_kernel(const __bf16* __restrict__ xb, const __bf16* __restrict__ wqkv,
                     const float* __restrict__ cosT, const float* __restrict__ sinT,
                     __bf16* __restrict__ qb, __bf16* __restrict__ kbuf,
                     __bf16* __restrict__ vb)
{
  __shared__ __align__(16) __bf16 As2[2][BM][BK + LP];
  __shared__ __align__(16) __bf16 Bt2[2][BN][BK + LP];
  const int lane = threadIdx.x & 31;
  const int half = lane >> 4;
  const int mr   = lane & 15;
  const int m0   = (threadIdx.x >> 5) * 16;
  const int row0 = blockIdx.x * BM;
  const int nb   = blockIdx.y;            // 0..47
  const int section = nb >> 4;            // 0=q,1=k,2=v
  const int head    = nb & 15;
  const int col0 = nb * BN;

  v8f acc[4];
  gemm_core_pipe(xb, wqkv, kHid, 3 * kHid, row0, col0, As2, Bt2, acc);

  CFrag cf[4];
  #pragma unroll
  for (int nt = 0; nt < 4; ++nt) cf[nt].v = acc[nt];

  __bf16* dst = (section == 0) ? qb : (section == 1) ? kbuf : vb;

  #pragma unroll
  for (int nt = 0; nt < 4; ++nt) {
    #pragma unroll
    for (int v = 0; v < 8; ++v) {
      int t = row0 + m0 + v + 8 * half;     // token
      int s = t & (kSeq - 1);
      int bi = t >> 10;
      int d = nt * 16 + mr;                 // 0..63 within head
      float val = cf[nt].f[v];
      if (section < 2) {
        float cv = cosT[s * kHD + d];
        float sv = sinT[s * kHD + d];
        float partner = cf[nt ^ 2].f[v];
        float rot = (d < 32) ? -partner : partner;
        val = val * cv + rot * sv;
      }
      dst[(((size_t)bi * kNH + head) * kSeq + s) * kHD + d] = (__bf16)val;
    }
  }
}

// ---------------------------------------------------------------------------
// Flash-style attention: one block = (64 q rows, head, batch), 4 waves.
// QK^T and P*V via WMMA; online softmax with shfl_xor row reductions.
// ---------------------------------------------------------------------------
__global__ __launch_bounds__(128)
void attn_kernel(const __bf16* __restrict__ qb, const __bf16* __restrict__ kb,
                 const __bf16* __restrict__ vb, __bf16* __restrict__ ctxb)
{
  __shared__ __align__(16) __bf16 Qs[64][kHD + LP];
  __shared__ __align__(16) __bf16 Ks[64][kHD + LP];
  __shared__ __align__(16) __bf16 Vt[kHD][64 + LP];
  __shared__ __align__(16) __bf16 Ps[4][16][64 + LP];

  const int tid  = threadIdx.x;
  const int wave = tid >> 5;
  const int lane = tid & 31;
  const int half = lane >> 4;
  const int mr   = lane & 15;
  const int m0   = wave * 16;

  const int q0 = blockIdx.x * 64;
  const int h  = blockIdx.y;
  const int bi = blockIdx.z;
  const size_t hb = ((size_t)bi * kNH + h) * kSeq * kHD;
  const __bf16* Qg = qb + hb;
  const __bf16* Kg = kb + hb;
  const __bf16* Vg = vb + hb;

  // Q tile (row-major): 64x64 bf16 = 512 x b128, 4 async copies per thread
  #pragma unroll
  for (int i = 0; i < 4; ++i) {
    int idx = tid + i * 128;
    int r = idx >> 3, c = idx & 7;
    async_copy_b128(&Qs[r][c * 8], Qg + (size_t)(q0 + r) * kHD + c * 8);
  }
  async_wait0();
  __syncthreads();

  AFrag qf[2];
  #pragma unroll
  for (int kk = 0; kk < 2; ++kk)
    #pragma unroll
    for (int v = 0; v < 8; ++v) {
      int off = kk * 32 + ((v & 4) << 2) + half * 8 + (v & 3) * 2;
      qf[kk].u[v] = *(const unsigned*)&Qs[m0 + mr][off];
    }

  CFrag oc[4];
  #pragma unroll
  for (int nt = 0; nt < 4; ++nt) oc[nt].v = zero_v8f();
  float mstat[8], lstat[8];
  #pragma unroll
  for (int v = 0; v < 8; ++v) { mstat[v] = -1e30f; lstat[v] = 0.0f; }

  for (int s0 = 0; s0 < kSeq; s0 += 64) {
    __syncthreads();
    #pragma unroll
    for (int i = 0; i < 4; ++i) {
      int idx = tid + i * 128;
      int r = idx >> 3, c = idx & 7;
      async_copy_b128(&Ks[r][c * 8], Kg + (size_t)(s0 + r) * kHD + c * 8);
    }
    for (int idx = tid; idx < 64 * 64; idx += 128) {
      int key = idx >> 6, d = idx & 63;
      Vt[d][key] = Vg[(size_t)(s0 + key) * kHD + d];
    }
    async_wait0();
    __syncthreads();

    // scores = Q * K^T  (B-frag element (k=hd, n=key) = Ks[key][hd])
    v8f sacc[4];
    #pragma unroll
    for (int nt = 0; nt < 4; ++nt) sacc[nt] = zero_v8f();
    #pragma unroll
    for (int kk = 0; kk < 2; ++kk) {
      AFrag bf[4];
      #pragma unroll
      for (int nt = 0; nt < 4; ++nt)
        #pragma unroll
        for (int v = 0; v < 8; ++v)
          bf[nt].u[v] = *(const unsigned*)&Ks[nt * 16 + mr][kk * 32 + ((v & 4) << 2) + half * 8 + (v & 3) * 2];
      #pragma unroll
      for (int nt = 0; nt < 4; ++nt)
        sacc[nt] = __builtin_amdgcn_wmma_f32_16x16x32_bf16(
            false, qf[kk].v, false, bf[nt].v, (short)0, sacc[nt], false, false);
    }
    CFrag sf[4];
    #pragma unroll
    for (int nt = 0; nt < 4; ++nt) {
      sf[nt].v = sacc[nt];
      #pragma unroll
      for (int v = 0; v < 8; ++v) sf[nt].f[v] *= 0.125f;   // HD^-0.5
    }

    // online softmax per row (row v+8*half; 16 key cols per tile across lanes)
    #pragma unroll
    for (int v = 0; v < 8; ++v) {
      float rmax = sf[0].f[v];
      #pragma unroll
      for (int nt = 1; nt < 4; ++nt) rmax = fmaxf(rmax, sf[nt].f[v]);
      #pragma unroll
      for (int m = 1; m < 16; m <<= 1) rmax = fmaxf(rmax, __shfl_xor(rmax, m, 32));
      float mnew = fmaxf(mstat[v], rmax);
      float corr = __expf(mstat[v] - mnew);
      float psum = 0.0f;
      #pragma unroll
      for (int nt = 0; nt < 4; ++nt) {
        float p = __expf(sf[nt].f[v] - mnew);
        sf[nt].f[v] = p;
        psum += p;
      }
      #pragma unroll
      for (int m = 1; m < 16; m <<= 1) psum += __shfl_xor(psum, m, 32);
      lstat[v] = lstat[v] * corr + psum;
      mstat[v] = mnew;
      #pragma unroll
      for (int nt = 0; nt < 4; ++nt) {
        oc[nt].f[v] *= corr;
        Ps[wave][v + 8 * half][nt * 16 + mr] = (__bf16)sf[nt].f[v];
      }
    }

    // O += P * V  (A-frag from per-wave Ps; B-frag (k=key,n=hd) = Vt[hd][key])
    #pragma unroll
    for (int kk = 0; kk < 2; ++kk) {
      AFrag pf;
      #pragma unroll
      for (int v = 0; v < 8; ++v) {
        int off = kk * 32 + ((v & 4) << 2) + half * 8 + (v & 3) * 2;
        pf.u[v] = *(const unsigned*)&Ps[wave][mr][off];
      }
      AFrag vf[4];
      #pragma unroll
      for (int nt = 0; nt < 4; ++nt)
        #pragma unroll
        for (int v = 0; v < 8; ++v)
          vf[nt].u[v] = *(const unsigned*)&Vt[nt * 16 + mr][kk * 32 + ((v & 4) << 2) + half * 8 + (v & 3) * 2];
      #pragma unroll
      for (int nt = 0; nt < 4; ++nt)
        oc[nt].v = __builtin_amdgcn_wmma_f32_16x16x32_bf16(
            false, pf.v, false, vf[nt].v, (short)0, oc[nt].v, false, false);
    }
  }

  #pragma unroll
  for (int nt = 0; nt < 4; ++nt)
    #pragma unroll
    for (int v = 0; v < 8; ++v) {
      int srow = q0 + m0 + v + 8 * half;
      int d = nt * 16 + mr;
      float inv = __builtin_amdgcn_rcpf(lstat[v]);
      ctxb[((size_t)bi * kSeq + srow) * kHid + h * kHD + d] = (__bf16)(oc[nt].f[v] * inv);
    }
}

// ---------------------------------------------------------------------------
// Small elementwise / reduction kernels
// ---------------------------------------------------------------------------
__global__ void f2b_kernel(const float* __restrict__ src, __bf16* __restrict__ dst,
                           size_t n) {
  size_t i = (size_t)blockIdx.x * 256 + threadIdx.x;
  if (i < n) dst[i] = (__bf16)src[i];
}

__global__ void zero_kernel(float* __restrict__ p, size_t n) {
  size_t i = (size_t)blockIdx.x * 256 + threadIdx.x;
  if (i < n) p[i] = 0.0f;
}

// out = rmsnorm(a [+ b]); optional f32 and bf16 outputs. One row per block.
__global__ __launch_bounds__(256)
void rmsnorm_kernel(const float* __restrict__ a, const float* __restrict__ b,
                    float* __restrict__ outf, __bf16* __restrict__ outb) {
  __shared__ float red[8];
  const int row = blockIdx.x;
  const int wave = threadIdx.x >> 5, lane = threadIdx.x & 31;
  float vals[4], ss = 0.0f;
  #pragma unroll
  for (int i = 0; i < 4; ++i) {
    int c = threadIdx.x + i * 256;
    float x = a[(size_t)row * kHid + c];
    if (b) x += b[(size_t)row * kHid + c];
    vals[i] = x; ss += x * x;
  }
  #pragma unroll
  for (int m = 1; m < 32; m <<= 1) ss += __shfl_xor(ss, m, 32);
  if (lane == 0) red[wave] = ss;
  __syncthreads();
  float tot = 0.0f;
  #pragma unroll
  for (int w = 0; w < 8; ++w) tot += red[w];
  float sc = rsqrtf(tot / (float)kHid + kEps);
  #pragma unroll
  for (int i = 0; i < 4; ++i) {
    int c = threadIdx.x + i * 256;
    float y = vals[i] * sc;
    if (outf) outf[(size_t)row * kHid + c] = y;
    if (outb) outb[(size_t)row * kHid + c] = (__bf16)y;
  }
}

// Gating: one wave per token; softmax over 8 experts, top-2 renormalized.
__global__ __launch_bounds__(32)
void gate_kernel(const float* __restrict__ hs, const float* __restrict__ wg,
                 float* __restrict__ gate_w) {
  const int t = blockIdx.x, lane = threadIdx.x;
  float acc[kE];
  #pragma unroll
  for (int e = 0; e < kE; ++e) acc[e] = 0.0f;
  for (int hcol = lane; hcol < kHid; hcol += 32) {
    float x = hs[(size_t)t * kHid + hcol];
    #pragma unroll
    for (int e = 0; e < kE; ++e) acc[e] += x * wg[hcol * kE + e];
  }
  #pragma unroll
  for (int m = 1; m < 32; m <<= 1)
    #pragma unroll
    for (int e = 0; e < kE; ++e) acc[e] += __shfl_xor(acc[e], m, 32);
  if (lane == 0) {
    float mx = acc[0];
    #pragma unroll
    for (int e = 1; e < kE; ++e) mx = fmaxf(mx, acc[e]);
    float p[kE], sum = 0.0f;
    #pragma unroll
    for (int e = 0; e < kE; ++e) { p[e] = __expf(acc[e] - mx); sum += p[e]; }
    float rs = __builtin_amdgcn_rcpf(sum);
    #pragma unroll
    for (int e = 0; e < kE; ++e) p[e] *= rs;
    int i1 = 0;
    #pragma unroll
    for (int e = 1; e < kE; ++e) if (p[e] > p[i1]) i1 = e;
    int i2 = (i1 == 0) ? 1 : 0;
    #pragma unroll
    for (int e = 0; e < kE; ++e) if (e != i1 && p[e] > p[i2]) i2 = e;
    float s = fmaxf(p[i1] + p[i2], 1e-8f);
    float inv = __builtin_amdgcn_rcpf(s);
    #pragma unroll
    for (int e = 0; e < kE; ++e) gate_w[(size_t)t * kE + e] = 0.0f;
    gate_w[(size_t)t * kE + i1] = p[i1] * inv;
    gate_w[(size_t)t * kE + i2] = p[i2] * inv;
  }
}

} // namespace

extern "C" void kernel_launch(void* const* d_in, const int* in_sizes, int n_in,
                              void* d_out, int out_size, void* d_ws, size_t ws_size,
                              hipStream_t stream) {
  (void)in_sizes; (void)n_in; (void)out_size; (void)ws_size;
  const float* x     = (const float*)d_in[0];
  const float* cosT  = (const float*)d_in[1];
  const float* sinT  = (const float*)d_in[2];
  const float* w_qkv = (const float*)d_in[3];
  const float* w_o   = (const float*)d_in[4];
  const float* w_g   = (const float*)d_in[5];
  const float* w_gu  = (const float*)d_in[6];
  const float* w_dn  = (const float*)d_in[7];
  float* out = (float*)d_out;

  char* ws = (char*)d_ws;
  size_t off = 0;
  auto take = [&](size_t bytes) -> void* {
    void* p = ws + off;
    off += (bytes + 255) & ~(size_t)255;
    return p;
  };

  __bf16* xb     = (__bf16*)take((size_t)kTok * kHid * 2);
  __bf16* wqkv_b = (__bf16*)take((size_t)kHid * 3 * kHid * 2);
  __bf16* wo_b   = (__bf16*)take((size_t)kHid * kHid * 2);
  __bf16* wgu_b  = (__bf16*)take((size_t)kE * kHid * kGU * 2);
  __bf16* wdn_b  = (__bf16*)take((size_t)kE * kInter * kHid * 2);
  __bf16* qb     = (__bf16*)take((size_t)kTok * kHid * 2);
  __bf16* kb     = (__bf16*)take((size_t)kTok * kHid * 2);
  __bf16* vb     = (__bf16*)take((size_t)kTok * kHid * 2);
  __bf16* ctxb   = (__bf16*)take((size_t)kTok * kHid * 2);
  float*  attn_o = (float*)take((size_t)kTok * kHid * 4);
  float*  hs     = (float*)take((size_t)kTok * kHid * 4);
  __bf16* hs_b   = (__bf16*)take((size_t)kTok * kHid * 2);
  float*  gatew  = (float*)take((size_t)kTok * kE * 4);
  __bf16* act_b  = (__bf16*)take((size_t)kTok * kInter * 2);
  float*  mixed  = (float*)take((size_t)kTok * kHid * 4);

  auto cdiv = [](size_t a, size_t b) { return (unsigned)((a + b - 1) / b); };

  // 1) fp32 -> bf16 conversions (activations + weights)
  f2b_kernel<<<cdiv((size_t)kTok * kHid, 256), 256, 0, stream>>>(x, xb, (size_t)kTok * kHid);
  f2b_kernel<<<cdiv((size_t)kHid * 3 * kHid, 256), 256, 0, stream>>>(w_qkv, wqkv_b, (size_t)kHid * 3 * kHid);
  f2b_kernel<<<cdiv((size_t)kHid * kHid, 256), 256, 0, stream>>>(w_o, wo_b, (size_t)kHid * kHid);
  f2b_kernel<<<cdiv((size_t)kE * kHid * kGU, 256), 256, 0, stream>>>(w_gu, wgu_b, (size_t)kE * kHid * kGU);
  f2b_kernel<<<cdiv((size_t)kE * kInter * kHid, 256), 256, 0, stream>>>(w_dn, wdn_b, (size_t)kE * kInter * kHid);

  // 2) QKV GEMM + fused RoPE
  qkv_rope_kernel<<<dim3(kTok / BM, (3 * kHid) / BN), 256, 0, stream>>>(
      xb, wqkv_b, cosT, sinT, qb, kb, vb);

  // 3) Flash attention
  attn_kernel<<<dim3(kSeq / 64, kNH, kBch), 128, 0, stream>>>(qb, kb, vb, ctxb);

  // 4) Output projection + residual: attn_o = x + ctx @ w_o
  gemm_bf16_kernel<1><<<dim3(kTok / BM, kHid / BN), 256, 0, stream>>>(
      ctxb, wo_b, attn_o, kHid, kHid, x, nullptr, 0);

  // 5) hs = rmsnorm(x + attn_out); also bf16 copy for MoE GEMMs
  rmsnorm_kernel<<<kTok, 256, 0, stream>>>(attn_o, nullptr, hs, hs_b);

  // 6) Gating weights (top-2 renormalized softmax)
  gate_kernel<<<kTok, 32, 0, stream>>>(hs, w_g, gatew);

  // 7) MoE: dense per expert; fused gate_up+SiLU, then weighted-accumulate down
  zero_kernel<<<cdiv((size_t)kTok * kHid, 256), 256, 0, stream>>>(mixed, (size_t)kTok * kHid);
  for (int e = 0; e < kE; ++e) {
    moe_gateup_silu_kernel<<<dim3(kTok / BM, kInter / BN), 256, 0, stream>>>(
        hs_b, wgu_b + (size_t)e * kHid * kGU, act_b);
    gemm_bf16_kernel<2><<<dim3(kTok / BM, kHid / BN), 256, 0, stream>>>(
        act_b, wdn_b + (size_t)e * kInter * kHid, mixed, kHid, kInter,
        nullptr, gatew + e, kE);
  }

  // 8) out = rmsnorm(hs + mixed)
  rmsnorm_kernel<<<kTok, 256, 0, stream>>>(hs, mixed, out, nullptr);
}
